// EGCN_H_70042326663891
// MI455X (gfx1250) — matrix operand order repaired and checked by reading, not compiled, not run on hardware
//
#include <hip/hip_runtime.h>
#include <hip/hip_bf16.h>
#include <math.h>

typedef __attribute__((ext_vector_type(16))) _Float16 v16h;
typedef __attribute__((ext_vector_type(8)))  float    v8f;
typedef __attribute__((ext_vector_type(4)))  float    v4f;

#define TT  8
#define NN  4096
#define FIN 128
#define FF1 128
#define FF2 64
#define KSPLIT 4

// ---------- small helper kernels (scalar f32, negligible cost) ----------

__global__ void copy_f32_k(const float* __restrict__ s, float* __restrict__ d, int n) {
  int i = blockIdx.x * blockDim.x + threadIdx.x;
  if (i < n) d[i] = s[i];
}

// scores = feats @ p / ||p||2
__global__ void scores_k(const float* __restrict__ feats, const float* __restrict__ p,
                         float* __restrict__ scores, int n, int f) {
  int i = blockIdx.x * blockDim.x + threadIdx.x;
  if (i >= n) return;
  float nrm = 0.f, dot = 0.f;
  const float* fr = feats + (size_t)i * f;
  for (int s = 0; s < f; ++s) { float ps = p[s]; nrm += ps * ps; dot += fr[s] * ps; }
  scores[i] = dot / sqrtf(nrm);
}

// exact top-k, descending, stable tie order (matches jax.lax.top_k).
__global__ void topk_k(const float* __restrict__ s, int n, int k,
                       int* __restrict__ idx, float* __restrict__ val) {
  for (int i = threadIdx.x; i < n; i += blockDim.x) {
    float si = s[i];
    int rank = 0;
    for (int j = 0; j < n; ++j) {
      float sj = s[j];
      rank += (sj > si) || (sj == si && j < i);
    }
    if (rank < k) { idx[rank] = i; val[rank] = si; }
  }
}

// z[r][c] = feats[idx[c]][r] * tanh(val[c])   (z is [rows x k])
__global__ void zbuild_k(const float* __restrict__ feats, const int* __restrict__ idx,
                         const float* __restrict__ val, float* __restrict__ z,
                         int rows, int k, int ldf) {
  int i = blockIdx.x * blockDim.x + threadIdx.x;
  if (i >= rows * k) return;
  int r = i / k, c = i - r * k;
  z[i] = feats[(size_t)idx[c] * ldf + r] * tanhf(val[c]);
}

// u = sigmoid(W0 z + U0 Q + B0); r = sigmoid(W1 z + U1 Q + B1); wz2b = W2 z + B2
__global__ void gates_k(const float* __restrict__ W, const float* __restrict__ U,
                        const float* __restrict__ B, const float* __restrict__ Q,
                        const float* __restrict__ z,
                        float* __restrict__ ug, float* __restrict__ rg,
                        float* __restrict__ wz2b, int R, int C) {
  int i = blockIdx.x * blockDim.x + threadIdx.x;
  if (i >= R * C) return;
  int r = i / C, c = i - r * C;
  float a0 = B[i], a1 = B[R * C + i], a2 = B[2 * R * C + i];
  const float* w0 = W + (size_t)r * R;
  const float* w1 = W + (size_t)(R + r) * R;
  const float* w2 = W + (size_t)(2 * R + r) * R;
  const float* u0 = U + (size_t)r * R;
  const float* u1 = U + (size_t)(R + r) * R;
  for (int s = 0; s < R; ++s) {
    float zs = z[s * C + c], qs = Q[s * C + c];
    a0 += w0[s] * zs + u0[s] * qs;
    a1 += w1[s] * zs + u1[s] * qs;
    a2 += w2[s] * zs;
  }
  ug[i]   = 1.f / (1.f + expf(-a0));
  rg[i]   = 1.f / (1.f + expf(-a1));
  wz2b[i] = a2;
}

// h = tanh(wz2b + U2 @ (r*Q)); Qn = (1-u)*Q + u*h
__global__ void newq_k(const float* __restrict__ U2, const float* __restrict__ Q,
                       const float* __restrict__ ug, const float* __restrict__ rg,
                       const float* __restrict__ wz2b, float* __restrict__ Qn,
                       int R, int C) {
  int i = blockIdx.x * blockDim.x + threadIdx.x;
  if (i >= R * C) return;
  int r = i / C, c = i - r * C;
  float h = wz2b[i];
  const float* u2 = U2 + (size_t)r * R;
  for (int s = 0; s < R; ++s) h += u2[s] * (rg[s * C + c] * Q[s * C + c]);
  h = tanhf(h);
  float uu = ug[i], q = Q[i];
  Qn[i] = (1.f - uu) * q + uu * h;
}

// XT[c][n] = (f16)( sum_s feats[n][s] * Q[s][c] )   -- X transposed, f16 for WMMA B
__global__ void xt_k(const float* __restrict__ feats, const float* __restrict__ Q,
                     _Float16* __restrict__ XT, int n, int f, int c) {
  int i = blockIdx.x * blockDim.x + threadIdx.x;
  if (i >= n * c) return;
  int col = i / n, node = i - col * n;
  const float* fr = feats + (size_t)node * f;
  const float* qc = Q + col;
  float acc = 0.f;
  for (int s = 0; s < f; ++s) acc += fr[s] * qc[(size_t)s * c];
  XT[(size_t)col * n + node] = (_Float16)acc;
}

// out[i] = act( sum_s part[s][i] )  -- fixed order => deterministic
__global__ void combine_k(const float* __restrict__ part, float* __restrict__ out,
                          size_t n, int nsplit, int relu) {
  size_t i = (size_t)blockIdx.x * blockDim.x + threadIdx.x;
  if (i >= n) return;
  float a = 0.f;
  for (int s = 0; s < nsplit; ++s) a += part[(size_t)s * n + i];
  out[i] = relu ? fmaxf(a, 0.f) : a;
}

// ---------- heavy kernel: part = A_f32[rows, kslice] @ X[kslice, cols] ----------
// A: [NN x K] f32 adjacency, converted f16 in regs, NON-TEMPORAL loads (streamed).
// BT: [C x K] f16 (X transposed): each B-fragment is one contiguous 32B load (L2-resident).
// Wave -> 16 rows x 64 cols; block = 4 waves; grid = (NN/64, C/64, nsplit).
__global__ __launch_bounds__(128) void big_gemm_wmma(const float* __restrict__ A,
                                                     const _Float16* __restrict__ BT,
                                                     float* __restrict__ out,
                                                     int K, int KL, int C, int relu) {
  const int lane = threadIdx.x & 31;
  const int wave = threadIdx.x >> 5;
  const int r0 = (blockIdx.x * 4 + wave) * 16;
  const int c0 = blockIdx.y * 64;
  const int kbeg = blockIdx.z * KL;
  const int l  = lane & 15;   // A: row in tile; B/D: col in tile
  const int hi = lane >> 4;   // K-half selector per ISA fragment layout

  v8f acc0 = {}, acc1 = {}, acc2 = {}, acc3 = {};

  // A fragment: lane holds row l, K = hi*16 + {0..15} (matches B)
  const float*    arow = A  + (size_t)(r0 + l) * K + kbeg + hi * 16;
  // B fragment: lane holds col l, halves at K = hi*16 + {0..15} (contiguous in XT row)
  const _Float16* bp0  = BT + (size_t)(c0 + l) * K + kbeg + hi * 16;
  const _Float16* bp1  = bp0 + (size_t)16 * K;
  const _Float16* bp2  = bp0 + (size_t)32 * K;
  const _Float16* bp3  = bp0 + (size_t)48 * K;

  for (int k0 = 0; k0 < KL; k0 += 32) {
    v4f x0 = __builtin_nontemporal_load((const v4f*)(arow + k0));
    v4f x1 = __builtin_nontemporal_load((const v4f*)(arow + k0 + 4));
    v4f x2 = __builtin_nontemporal_load((const v4f*)(arow + k0 + 8));
    v4f x3 = __builtin_nontemporal_load((const v4f*)(arow + k0 + 12));
    v16h a;
    a[0]  = (_Float16)x0[0]; a[1]  = (_Float16)x0[1]; a[2]  = (_Float16)x0[2]; a[3]  = (_Float16)x0[3];
    a[4]  = (_Float16)x1[0]; a[5]  = (_Float16)x1[1]; a[6]  = (_Float16)x1[2]; a[7]  = (_Float16)x1[3];
    a[8]  = (_Float16)x2[0]; a[9]  = (_Float16)x2[1]; a[10] = (_Float16)x2[2]; a[11] = (_Float16)x2[3];
    a[12] = (_Float16)x3[0]; a[13] = (_Float16)x3[1]; a[14] = (_Float16)x3[2]; a[15] = (_Float16)x3[3];

    v16h b0 = *(const v16h*)(bp0 + k0);
    v16h b1 = *(const v16h*)(bp1 + k0);
    v16h b2 = *(const v16h*)(bp2 + k0);
    v16h b3 = *(const v16h*)(bp3 + k0);

    acc0 = __builtin_amdgcn_wmma_f32_16x16x32_f16(false, a, false, b0, (short)0, acc0, false, false);
    acc1 = __builtin_amdgcn_wmma_f32_16x16x32_f16(false, a, false, b1, (short)0, acc1, false, false);
    acc2 = __builtin_amdgcn_wmma_f32_16x16x32_f16(false, a, false, b2, (short)0, acc2, false, false);
    acc3 = __builtin_amdgcn_wmma_f32_16x16x32_f16(false, a, false, b3, (short)0, acc3, false, false);
  }

  // D layout: lane col = l, VGPR g -> row hi*8 + g.  Partial slab for this K slice.
  float* oslab = out + (size_t)blockIdx.z * NN * C;
  size_t obase = (size_t)(r0 + hi * 8) * C + c0 + l;
  for (int g = 0; g < 8; ++g) {
    float v0 = acc0[g], v1 = acc1[g], v2 = acc2[g], v3 = acc3[g];
    if (relu) {
      v0 = fmaxf(v0, 0.f); v1 = fmaxf(v1, 0.f);
      v2 = fmaxf(v2, 0.f); v3 = fmaxf(v3, 0.f);
    }
    size_t o = obase + (size_t)g * C;
    oslab[o]      = v0;
    oslab[o + 16] = v1;
    oslab[o + 32] = v2;
    oslab[o + 48] = v3;
  }
}

// -------------------------------- driver --------------------------------

extern "C" void kernel_launch(void* const* d_in, const int* in_sizes, int n_in,
                              void* d_out, int out_size, void* d_ws, size_t ws_size,
                              hipStream_t stream) {
  const float* adj   = (const float*)d_in[0];
  const float* feats = (const float*)d_in[1];
  const float* p1    = (const float*)d_in[2];
  const float* p2    = (const float*)d_in[3];
  const float* W1    = (const float*)d_in[4];
  const float* U1    = (const float*)d_in[5];
  const float* B1    = (const float*)d_in[6];
  const float* iW1   = (const float*)d_in[7];
  const float* W2    = (const float*)d_in[8];
  const float* U2    = (const float*)d_in[9];
  const float* B2    = (const float*)d_in[10];
  const float* iW2   = (const float*)d_in[11];
  float* out = (float*)d_out;

  // workspace carve (256B aligned)
  char* ws = (char*)d_ws;
  auto carve = [&](size_t bytes) -> char* {
    char* p = ws;
    ws += (bytes + 255) & ~(size_t)255;
    return p;
  };
  float*    Q1[2]; Q1[0] = (float*)carve(FIN * FF1 * 4); Q1[1] = (float*)carve(FIN * FF1 * 4);
  float*    Q2[2]; Q2[0] = (float*)carve(FF1 * FF2 * 4); Q2[1] = (float*)carve(FF1 * FF2 * 4);
  float*    sc   = (float*)   carve(NN * 4);
  int*      idx  = (int*)     carve(FF1 * 4);
  float*    val  = (float*)   carve(FF1 * 4);
  float*    z    = (float*)   carve(FIN * FF1 * 4);
  float*    ug   = (float*)   carve(FIN * FF1 * 4);
  float*    rg   = (float*)   carve(FIN * FF1 * 4);
  float*    wzb  = (float*)   carve(FIN * FF1 * 4);
  _Float16* XT   = (_Float16*)carve((size_t)FF1 * NN * 2);
  float*    h1   = (float*)   carve((size_t)NN * FF1 * 4);
  float*    part = (float*)   carve((size_t)KSPLIT * NN * FF1 * 4);  // K-split partials

  // If workspace is too small for partials, fall back to the single-pass path.
  const int S = ((char*)ws - (char*)d_ws) <= (ptrdiff_t)ws_size ? KSPLIT : 1;

  // init evolving weights
  copy_f32_k<<<(FIN * FF1 + 255) / 256, 256, 0, stream>>>(iW1, Q1[0], FIN * FF1);
  copy_f32_k<<<(FF1 * FF2 + 255) / 256, 256, 0, stream>>>(iW2, Q2[0], FF1 * FF2);
  int c1 = 0, c2 = 0;

  // helper: big GEMM + optional combine; writes act(adjT @ X) into dst [NN x C]
  auto spmm = [&](const float* adjT, const _Float16* xT, float* dst, int C, int relu) {
    if (S == 1) {
      big_gemm_wmma<<<dim3(NN / 64, C / 64, 1), 128, 0, stream>>>(adjT, xT, dst, NN, NN, C, relu);
    } else {
      big_gemm_wmma<<<dim3(NN / 64, C / 64, S), 128, 0, stream>>>(adjT, xT, part, NN, NN / S, C, 0);
      size_t n = (size_t)NN * C;
      combine_k<<<(unsigned)((n + 255) / 256), 256, 0, stream>>>(part, dst, n, S, relu);
    }
  };

  for (int t = 0; t < TT; ++t) {
    const float* adjT   = adj   + (size_t)t * NN * NN;
    const float* featsT = feats + (size_t)t * NN * FIN;
    const float* p1t    = p1 + t * FIN;
    const float* p2t    = p2 + t * FF1;

    // ---------------- layer 1 cell + embedding ----------------
    scores_k<<<NN / 256, 256, 0, stream>>>(featsT, p1t, sc, NN, FIN);
    topk_k<<<1, 256, 0, stream>>>(sc, NN, FF1, idx, val);
    zbuild_k<<<(FIN * FF1 + 255) / 256, 256, 0, stream>>>(featsT, idx, val, z, FIN, FF1, FIN);
    gates_k<<<(FIN * FF1 + 255) / 256, 256, 0, stream>>>(W1, U1, B1, Q1[c1], z, ug, rg, wzb, FIN, FF1);
    newq_k<<<(FIN * FF1 + 255) / 256, 256, 0, stream>>>(U1 + 2 * FIN * FIN, Q1[c1], ug, rg, wzb,
                                                        Q1[1 - c1], FIN, FF1);
    c1 ^= 1;
    xt_k<<<((size_t)NN * FF1 + 255) / 256, 256, 0, stream>>>(featsT, Q1[c1], XT, NN, FIN, FF1);
    spmm(adjT, XT, h1, FF1, 1);

    // ---------------- layer 2 cell (evolution only) ----------------
    scores_k<<<NN / 256, 256, 0, stream>>>(h1, p2t, sc, NN, FF1);
    topk_k<<<1, 256, 0, stream>>>(sc, NN, FF2, idx, val);
    zbuild_k<<<(FF1 * FF2 + 255) / 256, 256, 0, stream>>>(h1, idx, val, z, FF1, FF2, FF1);
    gates_k<<<(FF1 * FF2 + 255) / 256, 256, 0, stream>>>(W2, U2, B2, Q2[c2], z, ug, rg, wzb, FF1, FF2);
    newq_k<<<(FF1 * FF2 + 255) / 256, 256, 0, stream>>>(U2 + 2 * FF1 * FF1, Q2[c2], ug, rg, wzb,
                                                        Q2[1 - c2], FF1, FF2);
    c2 ^= 1;

    // layer-2 embedding only feeds the output; compute it only at t = T-1
    if (t == TT - 1) {
      xt_k<<<((size_t)NN * FF2 + 255) / 256, 256, 0, stream>>>(h1, Q2[c2], XT, NN, FF1, FF2);
      spmm(adjT, XT, out, FF2, 0);
    }
  }
}
